// W8A8Linear_15753940041870
// MI455X (gfx1250) — 8-bit matrix operand order repaired and checked by reading, compiled, not run on hardware
//
#include <hip/hip_runtime.h>
#include <stdint.h>


typedef __attribute__((ext_vector_type(8))) int v8i;

#define BB      4
#define SS      2048
#define DIN_    2048
#define DOUT_   2048
#define M_TOTAL (BB * SS)

__device__ __forceinline__ int q1(float v, float inv) {
    int i = (int)rintf(v * inv);
    i = i < -128 ? -128 : (i > 127 ? 127 : i);
    return i & 255;
}
__device__ __forceinline__ int pack4f(float4 f, float inv) {
    return q1(f.x, inv) | (q1(f.y, inv) << 8) | (q1(f.z, inv) << 16) | (q1(f.w, inv) << 24);
}

__global__ __launch_bounds__(256) void quant_x_kernel(const float* __restrict__ x,
                                                      const float* __restrict__ x_scale,
                                                      int8_t* __restrict__ xq, int n16) {
    int t = blockIdx.x * 256 + threadIdx.x;
    if (t >= n16) return;
    float inv = 1.0f / x_scale[0];
    const float4* xp = (const float4*)x + (size_t)t * 4;
    float4 f0 = xp[0], f1 = xp[1], f2 = xp[2], f3 = xp[3];
    int4 o;
    o.x = pack4f(f0, inv);
    o.y = pack4f(f1, inv);
    o.z = pack4f(f2, inv);
    o.w = pack4f(f3, inv);
    ((int4*)xq)[t] = o;
}

__device__ __forceinline__ int pack4i(int4 w) {
    return (w.x & 255) | ((w.y & 255) << 8) | ((w.z & 255) << 16) | ((w.w & 255) << 24);
}

__global__ __launch_bounds__(256) void pack_w_kernel(const int* __restrict__ w,
                                                     int8_t* __restrict__ w8, int n16) {
    int t = blockIdx.x * 256 + threadIdx.x;
    if (t >= n16) return;
    const int4* wp = (const int4*)w + (size_t)t * 4;
    int4 i0 = wp[0], i1 = wp[1], i2 = wp[2], i3 = wp[3];
    int4 o;
    o.x = pack4i(i0);
    o.y = pack4i(i1);
    o.z = pack4i(i2);
    o.w = pack4i(i3);
    ((int4*)w8)[t] = o;
}

__global__ __launch_bounds__(256) void gemm_i8_kernel(const int8_t* __restrict__ A,
                                                      const int8_t* __restrict__ W,
                                                      const float*  __restrict__ w_scale,
                                                      const float*  __restrict__ bias,
                                                      const float*  __restrict__ x_scale,
                                                      float* __restrict__ out) {
    const int K = DIN_;
    const int lane = threadIdx.x & 31;
    const int wave = threadIdx.x >> 5;
    const int half = lane >> 4;
    const int l16  = lane & 15;

    const int wm = wave & 3;
    const int wn = wave >> 2;
    const int m_w = blockIdx.y * 128 + wm * 32;
    const int n_w = blockIdx.x * 64  + wn * 32;

    v8i acc[2][2] = {};

    const int8_t* a_base0 = A + (size_t)(m_w + l16) * K + half * 16;
    const int8_t* a_base1 = a_base0 + (size_t)16 * K;
    const int8_t* b_base0 = W + (size_t)(n_w + l16) * K + half * 16;
    const int8_t* b_base1 = b_base0 + (size_t)16 * K;

    for (int k0 = 0; k0 < K; k0 += 64) {
        v8i a[2], b[2];

        {
            const int8_t* p0 = a_base0 + k0;
            const int8_t* p1 = a_base1 + k0;
            int2 c0 = *(const int2*)(p0 +  0);
            int2 c1 = *(const int2*)(p0 +  8);
            int2 c2 = *(const int2*)(p0 + 32);
            int2 c3 = *(const int2*)(p0 + 40);
            a[0][0] = c0.x; a[0][1] = c0.y; a[0][2] = c1.x; a[0][3] = c1.y;
            a[0][4] = c2.x; a[0][5] = c2.y; a[0][6] = c3.x; a[0][7] = c3.y;
            int2 d0 = *(const int2*)(p1 +  0);
            int2 d1 = *(const int2*)(p1 +  8);
            int2 d2 = *(const int2*)(p1 + 32);
            int2 d3 = *(const int2*)(p1 + 40);
            a[1][0] = d0.x; a[1][1] = d0.y; a[1][2] = d1.x; a[1][3] = d1.y;
            a[1][4] = d2.x; a[1][5] = d2.y; a[1][6] = d3.x; a[1][7] = d3.y;
        }
        {
            const int8_t* p0 = b_base0 + k0;
            const int8_t* p1 = b_base1 + k0;
            int4 lo0 = *(const int4*)(p0);
            int4 hi0 = *(const int4*)(p0 + 32);
            b[0][0] = lo0.x; b[0][1] = lo0.y; b[0][2] = lo0.z; b[0][3] = lo0.w;
            b[0][4] = hi0.x; b[0][5] = hi0.y; b[0][6] = hi0.z; b[0][7] = hi0.w;
            int4 lo1 = *(const int4*)(p1);
            int4 hi1 = *(const int4*)(p1 + 32);
            b[1][0] = lo1.x; b[1][1] = lo1.y; b[1][2] = lo1.z; b[1][3] = lo1.w;
            b[1][4] = hi1.x; b[1][5] = hi1.y; b[1][6] = hi1.z; b[1][7] = hi1.w;
        }

        if (k0 + 64 < K) {
            __builtin_prefetch(a_base0 + k0 + 64, 0, 1);
            __builtin_prefetch(b_base0 + k0 + 64, 0, 1);
        }

        acc[0][0] = __builtin_amdgcn_wmma_i32_16x16x64_iu8(true, a[0], true, b[0], acc[0][0], false, false);
        acc[0][1] = __builtin_amdgcn_wmma_i32_16x16x64_iu8(true, a[0], true, b[1], acc[0][1], false, false);
        acc[1][0] = __builtin_amdgcn_wmma_i32_16x16x64_iu8(true, a[1], true, b[0], acc[1][0], false, false);
        acc[1][1] = __builtin_amdgcn_wmma_i32_16x16x64_iu8(true, a[1], true, b[1], acc[1][1], false, false);
    }

    const float xs = x_scale[0];
    for (int ni = 0; ni < 2; ++ni) {
        const int col = n_w + ni * 16 + l16;
        const float sc = xs * w_scale[col];
        const float bv = bias[col];
        for (int mi = 0; mi < 2; ++mi) {
            const int row0 = m_w + mi * 16 + half * 8;
            float* op = out + (size_t)row0 * DOUT_ + col;
            #pragma unroll
            for (int r = 0; r < 8; ++r)
                op[(size_t)r * DOUT_] = (float)acc[mi][ni][r] * sc + bv;
        }
    }
}

extern "C" void kernel_launch(void* const* d_in, const int* in_sizes, int n_in,
                              void* d_out, int out_size, void* d_ws, size_t ws_size,
                              hipStream_t stream) {
    const float* x       = (const float*)d_in[0];
    const int*   w_i32   = (const int*)d_in[1];
    const float* w_scale = (const float*)d_in[2];
    const float* bias    = (const float*)d_in[3];
    const float* x_scale = (const float*)d_in[4];
    float* out = (float*)d_out;

    int8_t* xq = (int8_t*)d_ws;
    int8_t* w8 = (int8_t*)d_ws + (size_t)M_TOTAL * DIN_;

    const int n16x = (M_TOTAL * DIN_) / 16;
    quant_x_kernel<<<n16x / 256, 256, 0, stream>>>(x, x_scale, xq, n16x);

    const int n16w = (DOUT_ * DIN_) / 16;
    pack_w_kernel<<<n16w / 256, 256, 0, stream>>>(w_i32, w8, n16w);

    dim3 grid(DOUT_ / 64, M_TOTAL / 128);
    gemm_i8_kernel<<<grid, 256, 0, stream>>>(xq, w8, w_scale, bias, x_scale, out);
}
